// Attention_26027501814474
// MI455X (gfx1250) — matrix operand order repaired and checked by reading, not compiled, not run on hardware
//
#include <hip/hip_runtime.h>

// ---------------- problem constants ----------------
#define BSZ    2
#define SEQ    1024
#define DIM    4096
#define NH     32
#define NKV    8
#define HD     128
#define MAXSEQ 4096
#define NQK    (NH * HD)            // 4096 q columns
#define NKN    (NKV * HD)           // 1024 k (and v) columns
#define NTOT   (NQK + 2 * NKN)      // 6144 fused qkv columns
#define MROWS  (BSZ * SEQ)          // 2048 GEMM rows
#define KSTEP  64                   // k-chunk staged per TDM transfer
#define BROWS  64                   // B-panel rows (N) per block
#define BPAD   72                   // LDS row stride in elems (144B: 32DW+4DW pad)

typedef __bf16 bf16_t;
typedef __attribute__((ext_vector_type(16))) __bf16 v16bf;
typedef __attribute__((ext_vector_type(8)))  __bf16 v8bf;
typedef __attribute__((ext_vector_type(4)))  __bf16 v4bf;
typedef __attribute__((ext_vector_type(8)))  float  v8f;
typedef __attribute__((ext_vector_type(4)))  unsigned int v4u;
typedef __attribute__((ext_vector_type(8)))  int v8i;
typedef __attribute__((ext_vector_type(4)))  int v4i;

#define WMMA_BF16(a, b, c) \
  __builtin_amdgcn_wmma_f32_16x16x32_bf16(false, (a), false, (b), (short)0, (c), false, false)

// A-operand (16x32 bf16, M x K): lane L holds row M=L%16; VGPR halves cover
// K = 8*(L/16)+0..7 (elems 0..7) and K = 16+8*(L/16)+0..7 (elems 8..15).
__device__ __forceinline__ v16bf load_a_tile(const bf16_t* base, int lda, int lane) {
  const int l16 = lane & 15, lg = lane >> 4;
  const bf16_t* p = base + (size_t)l16 * lda + 8 * lg;
  v8bf lo = *(const v8bf*)(p);
  v8bf hi = *(const v8bf*)(p + 16);
  return __builtin_shufflevector(lo, hi, 0, 1, 2, 3, 4, 5, 6, 7,
                                 8, 9, 10, 11, 12, 13, 14, 15);
}

// B-operand (32x16 bf16, K x N) with B[k][n] = src[n][k], src row-major,
// contraction contiguous: lane L = column N=L%16, K = 8*(L/16)+0..7 and 16+8*(L/16)+0..7.
__device__ __forceinline__ v16bf load_b_tile(const bf16_t* base, int ldk, int lane) {
  const int l16 = lane & 15, lg = lane >> 4;
  const bf16_t* p = base + (size_t)l16 * ldk + 8 * lg;
  v8bf lo = *(const v8bf*)(p);
  v8bf hi = *(const v8bf*)(p + 16);
  return __builtin_shufflevector(lo, hi, 0, 1, 2, 3, 4, 5, 6, 7,
                                 8, 9, 10, 11, 12, 13, 14, 15);
}

// ---- Tensor Data Mover: DMA a (tile_rows x KSTEP) bf16 tile (row-major,
// leading dim `ld` elems) into LDS with 4-DWORD padding every 32 DWORDs
// (LDS row stride becomes BPAD elems = 144B -> conflict-free B reads).
__device__ __forceinline__ void tdm_load_panel(unsigned lds_off, const bf16_t* gsrc,
                                               int rows_total, int ld) {
  const unsigned long long ga = (unsigned long long)(uintptr_t)gsrc;
  v4u g0;
  g0.x = 1u;                                           // count=1, user D#
  g0.y = lds_off;                                      // lds_addr (bytes)
  g0.z = (unsigned)(ga & 0xffffffffu);                 // global_addr[31:0]
  g0.w = (unsigned)((ga >> 32) & 0x01ffffffu) | (2u << 30);  // addr[56:32]|type=2
  v8i g1;
  g1[0] = (int)((1u << 16) |                           // data_size = 2B
                (1u << 20) |                           // pad_enable
                (4u << 22) |                           // pad_interval: 32 DWORDs
                (3u << 25));                           // pad_amount: 4 DWORDs
  g1[1] = (int)(((unsigned)ld & 0xffffu) << 16);                 // tensor_dim0 lo
  g1[2] = (int)((((unsigned)rows_total & 0xffffu) << 16) |       // tensor_dim1 lo
                (((unsigned)ld >> 16) & 0xffffu));               // tensor_dim0 hi
  g1[3] = (int)(((unsigned)KSTEP << 16) |                        // tile_dim0 = 64
                (((unsigned)rows_total >> 16) & 0xffffu));       // tensor_dim1 hi
  g1[4] = BROWS;                                                 // tile_dim1 = 64
  g1[5] = ld;                                                    // dim0_stride lo
  g1[6] = 0;
  g1[7] = 0;
  v4i z4 = (v4i)0;
#if __clang_major__ >= 23
  v8i z8 = (v8i)0;
  __builtin_amdgcn_tensor_load_to_lds(g0, g1, z4, z4, z8, 0);
#else
  __builtin_amdgcn_tensor_load_to_lds(g0, g1, z4, z4, 0);
#endif
}

// ---------------- fp32 -> bf16 convert (vectorized x4) ----------------
__global__ void cvt_f32_bf16(const float* __restrict__ in, bf16_t* __restrict__ out, int n4) {
  int i = blockIdx.x * blockDim.x + threadIdx.x;
  if (i < n4) {
    float4 v = ((const float4*)in)[i];
    v4bf o = { (bf16_t)v.x, (bf16_t)v.y, (bf16_t)v.z, (bf16_t)v.w };
    ((v4bf*)out)[i] = o;
  }
}

// ================= GEMM core: 128Mx64N per block, 32Mx64N per wave ==========
// B panel staged to LDS by wave 0 via TDM, double-buffered (TENSORcnt).
struct GemmAcc { v8f a0[4]; v8f a1[4]; };

__device__ __forceinline__ void gemm_mainloop(GemmAcc& g, const bf16_t* __restrict__ Amat,
                                              const bf16_t* __restrict__ Bmat,
                                              int rowsB, int m0, int n0,
                                              bf16_t* bpanel /* [2][BROWS*BPAD] */,
                                              int lane, int wid) {
#pragma unroll
  for (int t = 0; t < 4; ++t) { g.a0[t] = (v8f)0.0f; g.a1[t] = (v8f)0.0f; }

  const int nsteps = DIM / KSTEP;
  if (wid == 0)
    tdm_load_panel((unsigned)(uintptr_t)bpanel, Bmat + (size_t)n0 * DIM, rowsB, DIM);

  int cur = 0;
  for (int i = 0; i < nsteps; ++i) {
    const int k0 = i * KSTEP;
    __syncthreads();                       // everyone done reading buf cur^1
    if (wid == 0) {
      if (i + 1 < nsteps) {
        tdm_load_panel((unsigned)(uintptr_t)(bpanel + (cur ^ 1) * (BROWS * BPAD)),
                       Bmat + (size_t)n0 * DIM + (k0 + KSTEP), rowsB, DIM);
        __builtin_amdgcn_s_wait_tensorcnt(1);   // oldest (buf cur) complete
      } else {
        __builtin_amdgcn_s_wait_tensorcnt(0);
      }
    }
    __syncthreads();                       // buf cur visible to all waves
    const bf16_t* Bp = bpanel + cur * (BROWS * BPAD);
#pragma unroll
    for (int kk = 0; kk < KSTEP; kk += 32) {
      v16bf av0 = load_a_tile(Amat + (size_t)m0 * DIM + k0 + kk, DIM, lane);
      v16bf av1 = load_a_tile(Amat + (size_t)(m0 + 16) * DIM + k0 + kk, DIM, lane);
#pragma unroll
      for (int t = 0; t < 4; ++t) {
        v16bf bm = load_b_tile(Bp + (size_t)(16 * t) * BPAD + kk, BPAD, lane);
        g.a0[t] = WMMA_BF16(av0, bm, g.a0[t]);
        g.a1[t] = WMMA_BF16(av1, bm, g.a1[t]);
      }
    }
    cur ^= 1;
  }
}

// ---------------- fused QKV GEMM + RoPE + cache scatter ----------------
// grid (16, 96), block 128 (4 waves): rows bx*128 + wid*32, cols by*64.
__global__ __launch_bounds__(128) void qkv_kernel(
    const bf16_t* __restrict__ Xb,     // [2048][4096]
    const bf16_t* __restrict__ Wb,     // [6144][4096] rows: wq | wk | wv
    const float*  __restrict__ fcos,   // [1024][64]
    const float*  __restrict__ fsin,   // [1024][64]
    const int*    __restrict__ positions, // [1024]
    float*  __restrict__ cacheK,       // [2][4096][8][128] (output)
    float*  __restrict__ cacheV,       // [2][4096][8][128] (output)
    bf16_t* __restrict__ Qb,           // [2][32][1024][128]
    bf16_t* __restrict__ Kb,           // [2][8][1024][128]
    bf16_t* __restrict__ Vtb)          // [2][8][128][1024]  (transposed V)
{
  __shared__ __align__(128) bf16_t bpanel[2][BROWS * BPAD];  // 2 x 9 KB
  const int lane = threadIdx.x & 31;
  const int l16 = lane & 15, lg = lane >> 4;
  const int wid = threadIdx.x >> 5;
  const int m0 = blockIdx.x * 128 + wid * 32;
  const int n0 = blockIdx.y * 64;

  GemmAcc g;
  gemm_mainloop(g, Xb, Wb, NTOT, m0, n0, &bpanel[0][0], lane, wid);

#pragma unroll
  for (int half = 0; half < 2; ++half) {
    const int mbase = m0 + 16 * half;
#pragma unroll
    for (int t = 0; t < 4; ++t) {
      const int e0 = n0 + 16 * t;            // tile never straddles 4096/5120
      const int e = e0 + l16;
      const bool isQ = (e0 < NQK);
      const bool isK = (!isQ) && (e0 < NQK + NKN);
      const int d = e & (HD - 1);
      const int j = d >> 1;
#pragma unroll
      for (int r = 0; r < 8; ++r) {
        const int row = mbase + r + 8 * lg;  // C layout: M = r + 8*(lane/16)
        const int b = row >> 10;
        const int s = row & (SEQ - 1);
        float val = half ? g.a1[t][r] : g.a0[t][r];
        if (isQ || isK) {                    // RoPE: partner (re<->im) = lane^1
          float other = __shfl_xor(val, 1, 32);
          float c  = fcos[s * (HD / 2) + j];
          float sn = fsin[s * (HD / 2) + j];
          val = ((d & 1) == 0) ? (val * c - other * sn)
                               : (other * sn + val * c);
        }
        if (isQ) {
          const int h = e >> 7;
          Qb[(((size_t)b * NH + h) * SEQ + s) * HD + d] = (bf16_t)val;
        } else if (isK) {
          const int kvh = (e - NQK) >> 7;
          const int pos = positions[s] & (MAXSEQ - 1);
          cacheK[(((size_t)b * MAXSEQ + pos) * NKV + kvh) * HD + d] = val;
          Kb[(((size_t)b * NKV + kvh) * SEQ + s) * HD + d] = (bf16_t)val;
        } else {
          const int kvh = (e - NQK - NKN) >> 7;
          const int pos = positions[s] & (MAXSEQ - 1);
          cacheV[(((size_t)b * MAXSEQ + pos) * NKV + kvh) * HD + d] = val;
          Vtb[(((size_t)b * NKV + kvh) * HD + d) * SEQ + s] = (bf16_t)val;
        }
      }
    }
  }
}

// ---------------- flash attention: one wave per (b, h, 16-row Q tile) ------
__global__ __launch_bounds__(256) void attn_kernel(
    const bf16_t* __restrict__ Qb, const bf16_t* __restrict__ Kb,
    const bf16_t* __restrict__ Vtb, bf16_t* __restrict__ AOb) {
  __shared__ bf16_t lds_p[8][16 * 32]; // per-wave 16x32 bf16 P tile

  const int lane = threadIdx.x & 31;
  const int l16 = lane & 15, lg = lane >> 4;
  const int wid = threadIdx.x >> 5;
  const int gidx = blockIdx.x * 8 + wid;     // 0..4095
  const int qt = gidx & 63;
  const int h  = (gidx >> 6) & 31;
  const int b  = gidx >> 11;
  const int kv = h >> 2;                     // GQA repeat = 4
  const int q0 = qt * 16;

  const bf16_t* Qbase = Qb + (((size_t)b * NH + h) * SEQ + q0) * HD;
  const bf16_t* Kbase = Kb + ((size_t)b * NKV + kv) * SEQ * HD;
  const bf16_t* Vbase = Vtb + ((size_t)b * NKV + kv) * HD * SEQ;
  bf16_t* P = lds_p[wid];

  v16bf qa[4];
#pragma unroll
  for (int kk = 0; kk < 4; ++kk) qa[kk] = load_a_tile(Qbase + kk * 32, HD, lane);

  v8f o[8];
#pragma unroll
  for (int dt = 0; dt < 8; ++dt) o[dt] = (v8f)0.0f;
  float rmax[8], rsum[8];
#pragma unroll
  for (int r = 0; r < 8; ++r) { rmax[r] = -INFINITY; rsum[r] = 0.0f; }

  const float scale = 0.08838834764831845f;  // 1/sqrt(128)

  for (int k0 = 0; k0 < q0 + 16; k0 += 32) { // causal: only tiles with k<=q
    float pst[2][8];
#pragma unroll
    for (int nt = 0; nt < 2; ++nt) {
      v8f sc = (v8f)0.0f;
#pragma unroll
      for (int kk = 0; kk < 4; ++kk) {
        v16bf kb = load_b_tile(Kbase + (size_t)(k0 + nt * 16) * HD + kk * 32, HD, lane);
        sc = WMMA_BF16(qa[kk], kb, sc);
      }
      const int kcol = k0 + nt * 16 + l16;
#pragma unroll
      for (int r = 0; r < 8; ++r) {
        const int qrow = q0 + r + 8 * lg;
        pst[nt][r] = sc[r] * scale + (kcol <= qrow ? 0.0f : -1.0e9f);
      }
    }
#pragma unroll
    for (int r = 0; r < 8; ++r) {
      float tmax = fmaxf(pst[0][r], pst[1][r]);
#pragma unroll
      for (int off = 1; off < 16; off <<= 1) tmax = fmaxf(tmax, __shfl_xor(tmax, off, 32));
      const float nmax = fmaxf(rmax[r], tmax);
      const float corr = __expf(rmax[r] - nmax);   // first iter: exp(-inf)=0
      rmax[r] = nmax;
      const float p0 = __expf(pst[0][r] - nmax);
      const float p1 = __expf(pst[1][r] - nmax);
      pst[0][r] = p0; pst[1][r] = p1;
      float psum = p0 + p1;
#pragma unroll
      for (int off = 1; off < 16; off <<= 1) psum += __shfl_xor(psum, off, 32);
      rsum[r] = rsum[r] * corr + psum;
#pragma unroll
      for (int dt = 0; dt < 8; ++dt) o[dt][r] *= corr;
    }
    // C-layout -> A-layout transpose of P through per-wave LDS (same-wave
    // LDS ops are in-order; compiler inserts s_wait_dscnt for the read)
#pragma unroll
    for (int r = 0; r < 8; ++r) {
      const int m = r + 8 * lg;
      P[m * 32 + l16]      = (bf16_t)pst[0][r];
      P[m * 32 + 16 + l16] = (bf16_t)pst[1][r];
    }
    v16bf pa = load_a_tile(P, 32, lane);
#pragma unroll
    for (int dt = 0; dt < 8; ++dt) {           // O += P * V (V transposed)
      v16bf vb = load_b_tile(Vbase + (size_t)(dt * 16) * SEQ + k0, SEQ, lane);
      o[dt] = WMMA_BF16(pa, vb, o[dt]);
    }
  }

  bf16_t* Obase = AOb + ((size_t)b * SEQ + q0) * (NH * HD) + h * HD;
#pragma unroll
  for (int r = 0; r < 8; ++r) {
    const int m = r + 8 * lg;
    const float inv = 1.0f / rsum[r];
#pragma unroll
    for (int dt = 0; dt < 8; ++dt)
      Obase[(size_t)m * (NH * HD) + dt * 16 + l16] = (bf16_t)(o[dt][r] * inv);
  }
}

// ---------------- output projection GEMM (fp32 out) ----------------
// grid (16, 64), block 128 (4 waves), 32x64 per wave, TDM-staged B.
__global__ __launch_bounds__(128) void oproj_kernel(
    const bf16_t* __restrict__ AOb,    // [2048][4096]
    const bf16_t* __restrict__ Wob,    // [4096][4096]
    float* __restrict__ out)           // [2048][4096] fp32
{
  __shared__ __align__(128) bf16_t bpanel[2][BROWS * BPAD];
  const int lane = threadIdx.x & 31;
  const int l16 = lane & 15, lg = lane >> 4;
  const int wid = threadIdx.x >> 5;
  const int m0 = blockIdx.x * 128 + wid * 32;
  const int n0 = blockIdx.y * 64;

  GemmAcc g;
  gemm_mainloop(g, AOb, Wob, DIM, m0, n0, &bpanel[0][0], lane, wid);

#pragma unroll
  for (int half = 0; half < 2; ++half) {
    const int mbase = m0 + 16 * half;
#pragma unroll
    for (int t = 0; t < 4; ++t) {
      const int col = n0 + 16 * t + l16;
#pragma unroll
      for (int r = 0; r < 8; ++r) {
        const int row = mbase + r + 8 * lg;
        out[(size_t)row * DIM + col] = half ? g.a1[t][r] : g.a0[t][r];
      }
    }
  }
}

// ---------------- launcher ----------------
extern "C" void kernel_launch(void* const* d_in, const int* in_sizes, int n_in,
                              void* d_out, int out_size, void* d_ws, size_t ws_size,
                              hipStream_t stream) {
  const float* x     = (const float*)d_in[0];
  const int*   pos   = (const int*)d_in[1];
  // d_in[2] = mask (causal mask computed analytically)
  const float* wq    = (const float*)d_in[3];
  const float* wk    = (const float*)d_in[4];
  const float* wv    = (const float*)d_in[5];
  const float* wo    = (const float*)d_in[6];
  const float* fcos  = (const float*)d_in[7];
  const float* fsin  = (const float*)d_in[8];
  const float* cacheK_in = (const float*)d_in[9];
  const float* cacheV_in = (const float*)d_in[10];

  float* out        = (float*)d_out;
  float* cacheK_out = out + (size_t)BSZ * SEQ * DIM;
  float* cacheV_out = cacheK_out + (size_t)BSZ * MAXSEQ * NKV * HD;

  char* ws = (char*)d_ws;
  bf16_t* Xb   = (bf16_t*)ws;  ws += (size_t)MROWS * DIM * 2;
  bf16_t* Wqkv = (bf16_t*)ws;  ws += (size_t)NTOT * DIM * 2;
  bf16_t* Wob  = (bf16_t*)ws;  ws += (size_t)DIM * DIM * 2;
  bf16_t* Qb   = (bf16_t*)ws;  ws += (size_t)BSZ * NH * SEQ * HD * 2;
  bf16_t* Kb   = (bf16_t*)ws;  ws += (size_t)BSZ * NKV * SEQ * HD * 2;
  bf16_t* Vtb  = (bf16_t*)ws;  ws += (size_t)BSZ * NKV * HD * SEQ * 2;
  bf16_t* AOb  = (bf16_t*)ws;  ws += (size_t)MROWS * DIM * 2;

  const size_t cbytes = sizeof(float) * (size_t)BSZ * MAXSEQ * NKV * HD;
  hipMemcpyAsync(cacheK_out, cacheK_in, cbytes, hipMemcpyDeviceToDevice, stream);
  hipMemcpyAsync(cacheV_out, cacheV_in, cbytes, hipMemcpyDeviceToDevice, stream);

  auto cvt = [&](const float* src, bf16_t* dst, size_t n) {
    int n4 = (int)(n / 4);
    cvt_f32_bf16<<<(n4 + 255) / 256, 256, 0, stream>>>(src, dst, n4);
  };
  cvt(x,  Xb,                               (size_t)MROWS * DIM);
  cvt(wq, Wqkv,                             (size_t)NQK * DIM);
  cvt(wk, Wqkv + (size_t)NQK * DIM,         (size_t)NKN * DIM);
  cvt(wv, Wqkv + (size_t)(NQK + NKN) * DIM, (size_t)NKN * DIM);
  cvt(wo, Wob,                              (size_t)DIM * DIM);

  qkv_kernel<<<dim3(MROWS / 128, NTOT / 64), 128, 0, stream>>>(
      Xb, Wqkv, fcos, fsin, pos, cacheK_out, cacheV_out, Qb, Kb, Vtb);

  attn_kernel<<<(BSZ * NH * (SEQ / 16)) / 8, 256, 0, stream>>>(Qb, Kb, Vtb, AOb);

  oproj_kernel<<<dim3(MROWS / 128, DIM / 64), 128, 0, stream>>>(AOb, Wob, out);
}
